// SelfAttentionForHW_87376814669863
// MI455X (gfx1250) — matrix operand order repaired and checked by reading, not compiled, not run on hardware
//
#include <hip/hip_runtime.h>
#include <hip/hip_bf16.h>

// ---------------------------------------------------------------------------
// Self-attention energy+softmax for [B=8,C=512,T=16,W=32,H=32]
//   q = Wq·x + bq ; k = Wk·x + bk      (1x1x1 conv == channel GEMM)
//   energy[b] = Q·K^T  (M=N=K=1024)    -> softmax rows -> d_out [8,1024,1024]
// bf16 WMMA (v_wmma_f32_16x16x32_bf16) for both GEMMs, fp32 accumulate.
// Projection fuses Q and K so each x chunk is read from HBM exactly once
// (256 MB total, the roofline term at 23.3 TB/s), double-buffers its LDS
// staging, and batches all fragment loads before each WMMA chain.
// Attention kernel stages its Q strip with global_load_async_to_lds_b128
// (ASYNCcnt path) and fuses the row softmax.
// ---------------------------------------------------------------------------

typedef __attribute__((ext_vector_type(16))) __bf16        v16bf;
typedef __attribute__((ext_vector_type(8)))  float         v8f;
typedef __attribute__((ext_vector_type(8)))  unsigned int  v8u;

#define WSTR 520   // W LDS row stride (ushorts): 1040 B, 16B aligned
#define XSTR 40    // Xt LDS row stride (ushorts): 80 B, 16B aligned
#define XBUF (256 * XSTR)
#define QSTR 1032  // Q-strip LDS row stride (ushorts): 2064 B, 16B aligned
#define ESTR 1025  // energy LDS row stride (floats)

__device__ __forceinline__ unsigned int f2bf(float f) {
    unsigned int u = __float_as_uint(f);
    u += 0x7FFFu + ((u >> 16) & 1u);           // round-to-nearest-even
    return u >> 16;
}

__device__ __forceinline__ v16bf mkbf(uint4 lo, uint4 hi) {
    v8u t;
    t[0] = lo.x; t[1] = lo.y; t[2] = lo.z; t[3] = lo.w;
    t[4] = hi.x; t[5] = hi.y; t[6] = hi.z; t[7] = hi.w;
    return __builtin_bit_cast(v16bf, t);
}

// ---------------------------------------------------------------------------
// Kernel 1: fused Q+K projection GEMM.  grid = (4 n-blocks of 256, T=16, B=8)
// Each x chunk is staged once and feeds both the Wq and Wk WMMA chains.
// ---------------------------------------------------------------------------
__global__ __launch_bounds__(256)
void proj_bf16_kernel(const float* __restrict__ x,
                      const float* __restrict__ Wq, const float* __restrict__ bq,
                      const float* __restrict__ Wk, const float* __restrict__ bk,
                      unsigned short* __restrict__ qout,
                      unsigned short* __restrict__ kout)
{
    const int jblk = blockIdx.x;          // 256-column block of n
    const int t    = blockIdx.y;
    const int b    = blockIdx.z;

    extern __shared__ char smem[];
    unsigned short* Wl = (unsigned short*)smem;                     // [128][WSTR]
    unsigned short* Xt = (unsigned short*)(smem + 128 * WSTR * 2);  // 2x[256][XSTR]

    const int tid = threadIdx.x;
    const size_t xbase = (((size_t)b * 512) * 16 + t) * 1024 + jblk * 256;

    // Stage both weight matrices (rows 0..63 = Wq, 64..127 = Wk) as bf16.
    for (int idx = tid; idx < 128 * 256; idx += 256) {
        int r = idx >> 8, c0 = (idx & 255) * 2;
        const float* src = (r < 64) ? (Wq + (size_t)r * 512)
                                    : (Wk + (size_t)(r - 64) * 512);
        float2 wv = *(const float2*)(src + c0);
        *(unsigned int*)(Wl + r * WSTR + c0) = f2bf(wv.x) | (f2bf(wv.y) << 16);
    }

    // Stage first x chunk [32 c x 256 n] transposed -> Xt[0][n][c] (bf16).
    for (int idx = tid; idx < 16 * 256; idx += 256) {
        int c0 = (idx >> 8) * 2, n = idx & 255;
        size_t base = xbase + (size_t)c0 * 16 * 1024 + n;
        float v0 = x[base];
        float v1 = x[base + 16 * 1024];
        *(unsigned int*)(Xt + n * XSTR + c0) = f2bf(v0) | (f2bf(v1) << 16);
    }

    const int wv = tid >> 5, ln = tid & 31;
    const int it = wv >> 1;               // i-tile (cq block of 16): 0..3
    const int jh = wv & 1;                // which 128-col half of the 256 block
    const int m  = ln & 15, hh = ln >> 4; // WMMA lane coords

    v8f accq[8], acck[8];
    for (int jt = 0; jt < 8; ++jt)
        for (int e = 0; e < 8; ++e) { accq[jt][e] = 0.f; acck[jt][e] = 0.f; }

    __syncthreads();

    for (int step = 0; step < 16; ++step) {
        const int kk = step * 32;
        unsigned short* cur = Xt + (step & 1) * XBUF;
        unsigned short* nxt = Xt + ((step & 1) ^ 1) * XBUF;

        // 1) Load all fragments for this k-chunk into distinct registers.
        const unsigned short* apq = &Wl[(it * 16 + m) * WSTR + kk + 8 * hh];
        const unsigned short* apk = apq + 64 * WSTR;
        v16bf Aq = mkbf(*(const uint4*)apq, *(const uint4*)(apq + 16));
        v16bf Ak = mkbf(*(const uint4*)apk, *(const uint4*)(apk + 16));
        v16bf Bf[8];
        for (int jt = 0; jt < 8; ++jt) {
            const unsigned short* bp = &cur[((jh * 8 + jt) * 16 + m) * XSTR + 8 * hh];
            Bf[jt] = mkbf(*(const uint4*)bp, *(const uint4*)(bp + 16));
        }

        // 2) Stage next x chunk into the other buffer (overlaps with WMMAs).
        if (step < 15) {
            const size_t kbase = xbase + (size_t)(kk + 32) * 16 * 1024;
            for (int idx = tid; idx < 16 * 256; idx += 256) {
                int c0 = (idx >> 8) * 2, n = idx & 255;
                size_t base = kbase + (size_t)c0 * 16 * 1024 + n;
                if (step < 14)  // prefetch one chunk further ahead (HBM stream)
                    __builtin_prefetch(&x[base + (size_t)32 * 16 * 1024], 0, 1);
                float v0 = x[base];
                float v1 = x[base + 16 * 1024];
                *(unsigned int*)(nxt + n * XSTR + c0) = f2bf(v0) | (f2bf(v1) << 16);
            }
        }

        // 3) Two WMMA chains sharing the same B fragments.
        for (int jt = 0; jt < 8; ++jt)
            accq[jt] = __builtin_amdgcn_wmma_f32_16x16x32_bf16(
                false, Aq, false, Bf[jt], (short)0, accq[jt], false, false);
        for (int jt = 0; jt < 8; ++jt)
            acck[jt] = __builtin_amdgcn_wmma_f32_16x16x32_bf16(
                false, Ak, false, Bf[jt], (short)0, acck[jt], false, false);

        __syncthreads();
    }

    // D layout: lane gives n = m, rows = hh*8 + r.  Store as bf16, i = cq*16+t.
    for (int jt = 0; jt < 8; ++jt) {
        int jcol = jblk * 256 + (jh * 8 + jt) * 16 + m;
        for (int r = 0; r < 8; ++r) {
            int cq = it * 16 + hh * 8 + r;
            size_t o = ((size_t)b * 1024 + cq * 16 + t) * 1024 + jcol;
            qout[o] = (unsigned short)f2bf(accq[jt][r] + bq[cq]);
            kout[o] = (unsigned short)f2bf(acck[jt][r] + bk[cq]);
        }
    }
}

// ---------------------------------------------------------------------------
// Kernel 2: energy = Q·K^T (per-batch 1024x1024x1024) + fused row softmax.
// grid = (64 row-strips of 16, B=8), 256 threads (8 waves x 128 cols each).
// K is L2-resident (2 MB/batch vs 192 MB L2); B-frags stream from global.
// ---------------------------------------------------------------------------
__global__ __launch_bounds__(256)
void attn_softmax_kernel(const unsigned short* __restrict__ Q,
                         const unsigned short* __restrict__ K,
                         float* __restrict__ out)
{
    const int ib = blockIdx.x;   // 16-row strip index
    const int b  = blockIdx.y;

    extern __shared__ char smem[];
    unsigned short* Qs = (unsigned short*)smem;            // [16][QSTR]
    float*          E  = (float*)(smem + 16 * QSTR * 2);   // [16][ESTR]

    const int tid = threadIdx.x;
    const size_t qbase = ((size_t)b * 1024 + ib * 16) * 1024;

    // Stage the 16x1024 bf16 Q strip via the async-to-LDS path (ASYNCcnt).
    // Each lane issues one 16B async copy per iteration; no VGPR round-trip.
    for (int vi = tid; vi < 2048; vi += 256) {
        int r = vi >> 7, c8 = vi & 127;
        const unsigned short* gsrc = Q + qbase + (size_t)r * 1024 + c8 * 8;
        unsigned int ldsoff =
            (unsigned int)(size_t)(Qs + r * QSTR + c8 * 8);  // addr[31:0] = LDS offset
        asm volatile("global_load_async_to_lds_b128 %0, %1, off"
                     :: "v"(ldsoff), "v"(gsrc) : "memory");
    }
    asm volatile("s_wait_asynccnt 0x0" ::: "memory");
    __syncthreads();

    const int wv = tid >> 5, ln = tid & 31;
    const int m = ln & 15, hh = ln >> 4;
    const int jb = wv * 128;     // this wave's 128-column slab

    v8f acc[8];
    for (int jt = 0; jt < 8; ++jt)
        for (int e = 0; e < 8; ++e) acc[jt][e] = 0.f;

    const unsigned short* kbat = K + (size_t)b * 1024 * 1024;

    for (int kk = 0; kk < 1024; kk += 32) {
        // Batch all global B-frag loads + the LDS A-frag load up front.
        v16bf Bf[8];
        for (int jt = 0; jt < 8; ++jt) {
            const unsigned short* bp =
                kbat + (size_t)(jb + jt * 16 + m) * 1024 + kk + 8 * hh;
            Bf[jt] = mkbf(*(const uint4*)bp, *(const uint4*)(bp + 16));
        }
        const unsigned short* ap = Qs + m * QSTR + kk + 8 * hh;
        v16bf A = mkbf(*(const uint4*)ap, *(const uint4*)(ap + 16));

        for (int jt = 0; jt < 8; ++jt)
            acc[jt] = __builtin_amdgcn_wmma_f32_16x16x32_bf16(
                false, A, false, Bf[jt], (short)0, acc[jt], false, false);
    }

    // Dump accumulators into the fp32 energy strip.
    for (int jt = 0; jt < 8; ++jt)
        for (int r = 0; r < 8; ++r)
            E[(hh * 8 + r) * ESTR + jb + jt * 16 + m] = acc[jt][r];
    __syncthreads();

    // Fused softmax: 16 lanes per row, butterfly reductions inside wave32.
    const int row = tid >> 4, sub = tid & 15;
    float mx = -3.4e38f;
    for (int c = sub; c < 1024; c += 16)
        mx = fmaxf(mx, E[row * ESTR + c]);
    for (int off = 8; off; off >>= 1)
        mx = fmaxf(mx, __shfl_xor(mx, off, 16));

    float s = 0.f;
    for (int c = sub; c < 1024; c += 16) {
        float e = __expf(E[row * ESTR + c] - mx);
        E[row * ESTR + c] = e;
        s += e;
    }
    for (int off = 8; off; off >>= 1)
        s += __shfl_xor(s, off, 16);
    const float inv = 1.f / s;

    const size_t obase = ((size_t)b * 1024 + ib * 16 + row) * 1024;
    for (int c = sub; c < 1024; c += 16)
        out[obase + c] = E[row * ESTR + c] * inv;
}

// ---------------------------------------------------------------------------
extern "C" void kernel_launch(void* const* d_in, const int* in_sizes, int n_in,
                              void* d_out, int out_size, void* d_ws, size_t ws_size,
                              hipStream_t stream) {
    const float* x  = (const float*)d_in[0];
    const float* Wq = (const float*)d_in[1];
    const float* bq = (const float*)d_in[2];
    const float* Wk = (const float*)d_in[3];
    const float* bk = (const float*)d_in[4];
    float* out = (float*)d_out;

    unsigned short* Qb = (unsigned short*)d_ws;                 // [8][1024][1024] bf16
    unsigned short* Kb = Qb + (size_t)8 * 1024 * 1024;          // [8][1024][1024] bf16

    const size_t smem1 = (size_t)128 * WSTR * 2 + (size_t)2 * XBUF * 2;  // ~170 KB
    proj_bf16_kernel<<<dim3(4, 16, 8), 256, smem1, stream>>>(
        x, Wq, bq, Wk, bk, Qb, Kb);

    const size_t smem2 = (size_t)16 * QSTR * 2 + (size_t)16 * ESTR * 4;  // ~97 KB
    attn_softmax_kernel<<<dim3(64, 8), 256, smem2, stream>>>(Qb, Kb, out);
}
